// TransformerBlock_84516366450703
// MI455X (gfx1250) — matrix operand order repaired and checked by reading, not compiled, not run on hardware
//
#include <hip/hip_runtime.h>

// ---------------------------------------------------------------------------
// Types for CDNA5 WMMA (wave32): D(16x16 f32) = A(16x32 f16) * B(32x16 f16) + C
// ---------------------------------------------------------------------------
typedef _Float16 half_t;
typedef half_t v16h __attribute__((ext_vector_type(16)));
typedef half_t v8h  __attribute__((ext_vector_type(8)));
typedef float  v8f  __attribute__((ext_vector_type(8)));
typedef unsigned int u32x4 __attribute__((ext_vector_type(4)));
typedef int i32x4 __attribute__((ext_vector_type(4)));
typedef int i32x8 __attribute__((ext_vector_type(8)));

union AFrag { v16h v; v8h h[2]; };

#if defined(__gfx1250__) && __has_builtin(__builtin_amdgcn_tensor_load_to_lds)
#define USE_TDM 1
#pragma message("CDNA5 device pass: TDM (tensor_load_to_lds) path ENABLED")
#else
#define USE_TDM 0
#if defined(__gfx1250__)
#pragma message("CDNA5 device pass: TDM builtin missing -> cooperative fallback")
#endif
#endif

__device__ __forceinline__ v8f wmma_f16(v16h a, v16h b, v8f c) {
    // 8 args: (neg_a, A, neg_b, B, c_mod, C, reuse_a, reuse_b)
    return __builtin_amdgcn_wmma_f32_16x16x32_f16(false, a, false, b,
                                                  (short)0, c, false, false);
}

// 16-lane row max-reduction fused into DPP-modified VALU:
// v_max_num_f32_dpp dst, src0(rotated), src1  => rotate+max in one instruction.
// Rotation butterfly (1,2,4,8) within a 16-lane row fully reduces the row and
// leaves the result in every lane. s_delay_alu is a perf hint only (ISA ch.8),
// so the back-to-back DPP chain is functionally safe.
__device__ __forceinline__ float row_max_dpp(float v) {
    asm("v_max_num_f32_dpp %0, %0, %0 row_ror:1 row_mask:0xf bank_mask:0xf bound_ctrl:1\n\t"
        "v_max_num_f32_dpp %0, %0, %0 row_ror:2 row_mask:0xf bank_mask:0xf bound_ctrl:1\n\t"
        "v_max_num_f32_dpp %0, %0, %0 row_ror:4 row_mask:0xf bank_mask:0xf bound_ctrl:1\n\t"
        "v_max_num_f32_dpp %0, %0, %0 row_ror:8 row_mask:0xf bank_mask:0xf bound_ctrl:1"
        : "+v"(v));
    return v;
}

// A-fragment (16x32 f16) from a row-major [rows, ldk] source (global or LDS).
__device__ __forceinline__ v16h load_a_frag(const half_t* A, int row, int ldk,
                                            int kbase, int hk) {
    AFrag f;
    const half_t* p = A + (size_t)row * ldk + kbase + hk * 16;
    f.h[0] = *(const v8h*)(p);
    f.h[1] = *(const v8h*)(p + 8);
    return f.v;
}

// B-fragment (32x16 f16) from transposed source Bt[N,K]: lane = column,
// elems[0..15] = Bt[col, kbase + hk*16 + 0..15]
__device__ __forceinline__ v16h load_b_frag(const half_t* Bt, int col, int ldk,
                                            int kbase, int hk) {
    return *(const v16h*)(Bt + (size_t)col * ldk + kbase + hk * 16);
}

// ---------------------------------------------------------------------------
// TDM: issue a 2D tile load (rows x 32 f16, row stride = K elems) into LDS.
// D# packing per CDNA5 ISA section 8.3/8.4. One TDM op per call (wave-level).
// ---------------------------------------------------------------------------
#if USE_TDM
__device__ __forceinline__ void tdm_load_tile(unsigned lds_byte_off,
                                              const half_t* gsrc,
                                              unsigned K, unsigned rows) {
    unsigned long long ga = (unsigned long long)(size_t)gsrc;
    u32x4 g0;
    g0[0] = 1u;                                       // count=1, user descriptor
    g0[1] = lds_byte_off;                             // lds_addr (bytes)
    g0[2] = (unsigned)(ga & 0xFFFFFFFFu);             // global_addr[31:0]
    g0[3] = (unsigned)((ga >> 32) & 0x01FFFFFFu)      // global_addr[56:32]
          | 0x80000000u;                              // type=2 ("image")
    i32x8 g1;
    g1[0] = 0x00010000;                               // wg_mask=0, data_size=2B
    g1[1] = (int)((K & 0xFFFFu) << 16);               // tensor_dim0[15:0]
    g1[2] = (int)((K >> 16) | ((rows & 0xFFFFu) << 16)); // dim0 hi | tensor_dim1 lo
    g1[3] = (int)(32u << 16);                         // tensor_dim1 hi | tile_dim0=32
    g1[4] = (int)(rows & 0xFFFFu);                    // tile_dim1=rows, tile_dim2=0
    g1[5] = (int)K;                                   // tensor_dim0_stride[31:0]
    g1[6] = 0;                                        // stride hi | dim1_stride lo
    g1[7] = 0;
    i32x4 z4 = {0, 0, 0, 0};
#if __has_include(<hip/amd_detail/amd_gfx1250_TDM.h>)
    i32x8 z8 = {0, 0, 0, 0, 0, 0, 0, 0};
    __builtin_amdgcn_tensor_load_to_lds(g0, g1, z4, z4, z8, 0);   // 6-arg toolchain
#else
    __builtin_amdgcn_tensor_load_to_lds(g0, g1, z4, z4, 0);       // 5-arg toolchain
#endif
}

__device__ __forceinline__ void tdm_wait() {
#if __has_builtin(__builtin_amdgcn_s_wait_tensorcnt)
    __builtin_amdgcn_s_wait_tensorcnt(0);
#else
    asm volatile("s_wait_tensorcnt 0" ::: "memory");
#endif
}
#else
// Cooperative global->LDS staging fallback (also used by the host pass).
__device__ __forceinline__ void stage_coop(half_t* dst, const half_t* src,
                                           int rows, int K, int tid) {
    const int units = rows * 4;                       // 16B units of a rows x 32 tile
    for (int u = tid; u < units; u += 128) {
        int r = u >> 2, cc = (u & 3) * 8;
        *(v8h*)(dst + r * 32 + cc) = *(const v8h*)(src + (size_t)r * K + cc);
    }
}
#endif

// ---------------------------------------------------------------------------
// Weight transpose + f32->f16: Wt[n*K+k] = (f16)W[k*N+n].
// 32x32 LDS tile (padded), coalesced on both sides. Grid: (N/32, K/32).
// ---------------------------------------------------------------------------
__global__ __launch_bounds__(256)
void k_wtrans(const float* __restrict__ W, half_t* __restrict__ Wt, int K, int N) {
    __shared__ half_t tile[32][33];
    const int bn = blockIdx.x * 32;
    const int bk = blockIdx.y * 32;
    const int tx = threadIdx.x & 31;
    const int ty = threadIdx.x >> 5;
#pragma unroll
    for (int i = 0; i < 4; i++) {
        int k = bk + ty + i * 8;
        tile[ty + i * 8][tx] = (half_t)W[(size_t)k * N + bn + tx];
    }
    __syncthreads();
#pragma unroll
    for (int i = 0; i < 4; i++) {
        int n = bn + ty + i * 8;
        Wt[(size_t)n * K + bk + tx] = tile[tx][ty + i * 8];
    }
}

// ---------------------------------------------------------------------------
// LayerNorm over D=1024 per token, f32 in -> f16 out. One block per token.
// ---------------------------------------------------------------------------
__global__ __launch_bounds__(256)
void k_layernorm(const float* __restrict__ x, const float* __restrict__ g,
                 const float* __restrict__ be, half_t* __restrict__ out) {
    __shared__ float red[8];
    const size_t t = blockIdx.x;
    const float* xp = x + t * 1024;
    float v[4];
    float s = 0.f;
#pragma unroll
    for (int i = 0; i < 4; i++) { v[i] = xp[threadIdx.x + 256 * i]; s += v[i]; }
#pragma unroll
    for (int m = 1; m < 32; m <<= 1) s += __shfl_xor(s, m, 32);
    int wid = threadIdx.x >> 5;
    if ((threadIdx.x & 31) == 0) red[wid] = s;
    __syncthreads();
    float mean = 0.f;
#pragma unroll
    for (int i = 0; i < 8; i++) mean += red[i];
    mean *= (1.0f / 1024.0f);
    __syncthreads();
    float vs = 0.f;
#pragma unroll
    for (int i = 0; i < 4; i++) { float d = v[i] - mean; vs += d * d; }
#pragma unroll
    for (int m = 1; m < 32; m <<= 1) vs += __shfl_xor(vs, m, 32);
    if ((threadIdx.x & 31) == 0) red[wid] = vs;
    __syncthreads();
    float var = 0.f;
#pragma unroll
    for (int i = 0; i < 8; i++) var += red[i];
    var *= (1.0f / 1024.0f);
    float rinv = rsqrtf(var + 1e-5f);
    half_t* op = out + t * 1024;
#pragma unroll
    for (int i = 0; i < 4; i++) {
        int c = threadIdx.x + 256 * i;
        op[c] = (half_t)((v[i] - mean) * rinv * g[c] + be[c]);
    }
}

// ---------------------------------------------------------------------------
// WMMA GEMM with TDM double-buffered LDS staging.
// C[M,N] = A[M,K](f16) x Bt[N,K](f16)^T + bias, fused epilogue.
// Block = 128 threads = 4 waves (2x2); wave tile 32x64; block tile 64x128.
// Per k-step: wave 0 issues TDM loads of next A(64x32)/B(128x32) tiles while
// all waves compute 8 WMMAs from the current LDS buffers.
// MODE 0: out f32 = res + acc + bias                       (residual add)
// MODE 1: out f16 = relu(acc + bias), row-major [M,N]      (FFN hidden)
// MODE 2: out f16 = (acc + bias)*scale -> [B,H,S,Dk]       (Q with 1/sqrt(Dk), K)
// MODE 3: out f16 = acc + bias         -> [B,H,Dk,S]       (V transposed)
// ---------------------------------------------------------------------------
template <int MODE>
__device__ __forceinline__ void gemm_epilogue(v8f cc, int rbase, int cidx, int N,
                                              const float* __restrict__ bias,
                                              const float* __restrict__ res,
                                              void* __restrict__ out, float scale) {
    float bcol = bias[cidx];
#pragma unroll
    for (int r = 0; r < 8; r++) {
        int row = rbase + r;
        float v = cc[r] + bcol;
        if (MODE == 0) {
            float* o = (float*)out;
            o[(size_t)row * N + cidx] = res[(size_t)row * N + cidx] + v;
        } else if (MODE == 1) {
            half_t* o = (half_t*)out;
            o[(size_t)row * N + cidx] = (half_t)fmaxf(v, 0.f);
        } else if (MODE == 2) {
            half_t* o = (half_t*)out;
            v *= scale;
            int b = row >> 11, s = row & 2047;      // 2048 tokens per batch
            int h = cidx >> 6, dk = cidx & 63;      // 64 dims per head
            o[((size_t)(b * 16 + h) * 2048 + s) * 64 + dk] = (half_t)v;
        } else {
            half_t* o = (half_t*)out;
            int b = row >> 11, s = row & 2047;
            int h = cidx >> 6, dk = cidx & 63;
            o[((size_t)(b * 16 + h) * 64 + dk) * 2048 + s] = (half_t)v;
        }
    }
}

template <int MODE>
__global__ __launch_bounds__(128)
void k_gemm(const half_t* __restrict__ A, const half_t* __restrict__ Bt,
            const float* __restrict__ bias, const float* __restrict__ res,
            void* __restrict__ out, int M, int N, int K, float scale) {
    // smem element offsets: A buffers at 0 / 2048, B buffers at 4096 / 8192
    __shared__ half_t smem[12288];                   // 24 KB
    const int tid  = threadIdx.x;
    const int lane = tid & 31;
    const int w    = tid >> 5;
    const int wm = w >> 1, wn = w & 1;
    const int mblk = blockIdx.y * 64;
    const int nblk = blockIdx.x * 128;
    const int mb = mblk + wm * 32;
    const int nb = nblk + wn * 64;
    const int hk = lane >> 4, ln = lane & 15;

    const half_t* Ablk = A  + (size_t)mblk * K;
    const half_t* Bblk = Bt + (size_t)nblk * K;

    v8f c[2][4];
#pragma unroll
    for (int i = 0; i < 2; i++)
#pragma unroll
        for (int j = 0; j < 4; j++) c[i][j] = (v8f){};

    // ---- prologue: stage k=0 into buffer 0 ----
#if USE_TDM
    if (w == 0) {
        tdm_load_tile(0u,     Ablk, (unsigned)K, 64u);
        tdm_load_tile(8192u,  Bblk, (unsigned)K, 128u);
        tdm_wait();
    }
#else
    stage_coop(smem + 0,    Ablk, 64,  K, tid);
    stage_coop(smem + 4096, Bblk, 128, K, tid);
#endif
    __syncthreads();

    int buf = 0;
    for (int k = 0; k < K; k += 32) {
        // ---- issue next stage into the other buffer (overlaps compute) ----
        if (k + 32 < K) {
#if USE_TDM
            if (w == 0) {
                tdm_load_tile((unsigned)((buf ^ 1) * 4096),        Ablk + k + 32, (unsigned)K, 64u);
                tdm_load_tile((unsigned)(8192 + (buf ^ 1) * 8192), Bblk + k + 32, (unsigned)K, 128u);
            }
#else
            stage_coop(smem + (buf ^ 1) * 2048,            Ablk + k + 32, 64,  K, tid);
            stage_coop(smem + 4096 + (buf ^ 1) * 4096,     Bblk + k + 32, 128, K, tid);
#endif
        }
        // ---- compute from current buffers (LDS) ----
        const half_t* sA = smem + buf * 2048;
        const half_t* sB = smem + 4096 + buf * 4096;
        v16h a0 = load_a_frag(sA, wm * 32 +      ln, 32, 0, hk);
        v16h a1 = load_a_frag(sA, wm * 32 + 16 + ln, 32, 0, hk);
        v16h b0 = load_b_frag(sB, wn * 64 +  0 + ln, 32, 0, hk);
        v16h b1 = load_b_frag(sB, wn * 64 + 16 + ln, 32, 0, hk);
        v16h b2 = load_b_frag(sB, wn * 64 + 32 + ln, 32, 0, hk);
        v16h b3 = load_b_frag(sB, wn * 64 + 48 + ln, 32, 0, hk);
        c[0][0] = wmma_f16(a0, b0, c[0][0]);
        c[0][1] = wmma_f16(a0, b1, c[0][1]);
        c[0][2] = wmma_f16(a0, b2, c[0][2]);
        c[0][3] = wmma_f16(a0, b3, c[0][3]);
        c[1][0] = wmma_f16(a1, b0, c[1][0]);
        c[1][1] = wmma_f16(a1, b1, c[1][1]);
        c[1][2] = wmma_f16(a1, b2, c[1][2]);
        c[1][3] = wmma_f16(a1, b3, c[1][3]);
#if USE_TDM
        if (w == 0 && k + 32 < K) tdm_wait();
#endif
        __syncthreads();
        buf ^= 1;
    }
#pragma unroll
    for (int i = 0; i < 2; i++)
#pragma unroll
        for (int j = 0; j < 4; j++)
            gemm_epilogue<MODE>(c[i][j], mb + i * 16 + 8 * hk, nb + j * 16 + ln,
                                N, bias, res, out, scale);
    (void)M;
}

// ---------------------------------------------------------------------------
// Flash attention (no mask). Grid: (S/64, B*H). Block = 128 thr = 4 waves.
// Each wave: 16 query rows, streams keys in chunks of 32 with online softmax.
// Row max via fused DPP max-rotates (VALU only); row sums via P x ones WMMA.
// Q pre-scaled by 1/sqrt(Dk). K is [B,H,S,Dk] (== Bt layout for QK^T).
// V is [B,H,Dk,S] (== Bt layout for P*V). ctx out: [B,S,D] f16.
// ---------------------------------------------------------------------------
__global__ __launch_bounds__(128)
void k_attn(const half_t* __restrict__ Q, const half_t* __restrict__ Kh,
            const half_t* __restrict__ Vt, half_t* __restrict__ ctx) {
    __shared__ half_t lds_p[4][16 * 32];   // per-wave P bounce buffer (C->A layout)
    const int lane = threadIdx.x & 31;
    const int w    = threadIdx.x >> 5;
    const int hk = lane >> 4, ln = lane & 15;
    const int bh = blockIdx.y;
    const int qbase = blockIdx.x * 64 + w * 16;

    const half_t* Qb = Q  + (size_t)bh * 2048 * 64;
    const half_t* Kb = Kh + (size_t)bh * 2048 * 64;
    const half_t* Vb = Vt + (size_t)bh * 64 * 2048;

    const int qrow = qbase + ln;
    const v16h a0 = load_a_frag(Qb, qrow, 64, 0,  hk);   // dims 0..31
    const v16h a1 = load_a_frag(Qb, qrow, 64, 32, hk);   // dims 32..63

    v16h ones;
#pragma unroll
    for (int i = 0; i < 16; i++) ones[i] = (half_t)1.0f;

    v8f o0 = {}, o1 = {}, o2 = {}, o3 = {};
    float m[8], l[8], alpha[8];
#pragma unroll
    for (int r = 0; r < 8; r++) { m[r] = -3.0e38f; l[r] = 0.f; }

    half_t* pb = &lds_p[w][0];

    for (int kb = 0; kb < 2048; kb += 32) {
        // ---- scores: S[16q x 32k] = Q * K^T (two 16-key tiles) ----
        v16h bk00 = load_b_frag(Kb, kb +      ln, 64, 0,  hk);
        v16h bk01 = load_b_frag(Kb, kb +      ln, 64, 32, hk);
        v16h bk10 = load_b_frag(Kb, kb + 16 + ln, 64, 0,  hk);
        v16h bk11 = load_b_frag(Kb, kb + 16 + ln, 64, 32, hk);
        v8f s0 = {}, s1 = {};
        s0 = wmma_f16(a0, bk00, s0);
        s0 = wmma_f16(a1, bk01, s0);
        s1 = wmma_f16(a0, bk10, s1);
        s1 = wmma_f16(a1, bk11, s1);

        // ---- online softmax (rows = r + 8*hk): DPP row-max, exp, stage P ----
#pragma unroll
        for (int r = 0; r < 8; r++) {
            float mx = row_max_dpp(fmaxf(s0[r], s1[r]));
            float mnew = fmaxf(m[r], mx);
            alpha[r] = __expf(m[r] - mnew);
            m[r] = mnew;
            float p0 = __expf(s0[r] - mnew);
            float p1 = __expf(s1[r] - mnew);
            int prow = r + 8 * hk;
            pb[prow * 32 +      ln] = (half_t)p0;
            pb[prow * 32 + 16 + ln] = (half_t)p1;
            o0[r] *= alpha[r]; o1[r] *= alpha[r];
            o2[r] *= alpha[r]; o3[r] *= alpha[r];
        }
        asm volatile("s_wait_dscnt 0" ::: "memory");   // P writes visible to reads

        // ---- re-read P as an A-fragment (16x32, contraction = key) ----
        AFrag pf;
        pf.h[0] = *(const v8h*)&pb[ln * 32 + hk * 16];
        pf.h[1] = *(const v8h*)&pb[ln * 32 + hk * 16 + 8];
        v16h ap = pf.v;

        // ---- row sums of P via WMMA against ones: rs[r] broadcast to row ----
        v8f rs = {};
        rs = wmma_f16(ap, ones, rs);
#pragma unroll
        for (int r = 0; r < 8; r++) l[r] = l[r] * alpha[r] + rs[r];

        // ---- ctx accumulate: O[16q x 64d] += P * V (V transposed => Bt) ----
        v16h bv0 = *(const v16h*)(Vb + (size_t)( 0 + ln) * 2048 + kb + hk * 16);
        v16h bv1 = *(const v16h*)(Vb + (size_t)(16 + ln) * 2048 + kb + hk * 16);
        v16h bv2 = *(const v16h*)(Vb + (size_t)(32 + ln) * 2048 + kb + hk * 16);
        v16h bv3 = *(const v16h*)(Vb + (size_t)(48 + ln) * 2048 + kb + hk * 16);
        o0 = wmma_f16(ap, bv0, o0);
        o1 = wmma_f16(ap, bv1, o1);
        o2 = wmma_f16(ap, bv2, o2);
        o3 = wmma_f16(ap, bv3, o3);
    }

    // ---- normalize and store ctx [B,S,D] f16 ----
    const int b = bh >> 4, h = bh & 15;
#pragma unroll
    for (int r = 0; r < 8; r++) {
        float inv = 1.0f / l[r];
        int row = qbase + r + 8 * hk;
        size_t base = ((size_t)b * 2048 + row) * 1024 + h * 64 + ln;
        ctx[base +  0] = (half_t)(o0[r] * inv);
        ctx[base + 16] = (half_t)(o1[r] * inv);
        ctx[base + 32] = (half_t)(o2[r] * inv);
        ctx[base + 48] = (half_t)(o3[r] * inv);
    }
}

// ---------------------------------------------------------------------------
// Launch: full pre-LN transformer block.
// ---------------------------------------------------------------------------
extern "C" void kernel_launch(void* const* d_in, const int* in_sizes, int n_in,
                              void* d_out, int out_size, void* d_ws, size_t ws_size,
                              hipStream_t stream) {
    (void)in_sizes; (void)n_in; (void)out_size; (void)ws_size;
    const float* x     = (const float*)d_in[0];
    const float* ln1_g = (const float*)d_in[1];
    const float* ln1_b = (const float*)d_in[2];
    const float* ln2_g = (const float*)d_in[3];
    const float* ln2_b = (const float*)d_in[4];
    const float* Wq = (const float*)d_in[5];   const float* bq = (const float*)d_in[6];
    const float* Wk = (const float*)d_in[7];   const float* bk = (const float*)d_in[8];
    const float* Wv = (const float*)d_in[9];   const float* bv = (const float*)d_in[10];
    const float* Wo = (const float*)d_in[11];  const float* bo = (const float*)d_in[12];
    const float* W1 = (const float*)d_in[13];  const float* b1 = (const float*)d_in[14];
    const float* W2 = (const float*)d_in[15];  const float* b2 = (const float*)d_in[16];

    char* ws = (char*)d_ws;
    const size_t MB = 1024 * 1024;
    half_t* WqT = (half_t*)(ws +   0 * MB);   // [1024,1024] f16 (transposed)
    half_t* WkT = (half_t*)(ws +   2 * MB);
    half_t* WvT = (half_t*)(ws +   4 * MB);
    half_t* WoT = (half_t*)(ws +   6 * MB);
    half_t* W1T = (half_t*)(ws +   8 * MB);   // [4096,1024] f16
    half_t* W2T = (half_t*)(ws +  16 * MB);   // [1024,4096] f16
    half_t* Xn1 = (half_t*)(ws +  24 * MB);   // [8192,1024] f16
    half_t* Qh  = (half_t*)(ws +  40 * MB);   // [B,H,S,Dk] f16 (pre-scaled)
    half_t* Kh  = (half_t*)(ws +  56 * MB);   // [B,H,S,Dk] f16
    half_t* VtT = (half_t*)(ws +  72 * MB);   // [B,H,Dk,S] f16
    half_t* Ctx = (half_t*)(ws +  88 * MB);   // [B,S,D]    f16
    float*  X1  = (float*)(ws + 104 * MB);    // [8192,1024] f32 (post-attn residual)
    half_t* Xn2 = (half_t*)(ws + 136 * MB);   // [8192,1024] f16
    half_t* Hh  = (half_t*)(ws + 152 * MB);   // [8192,4096] f16

    // 1) weights -> transposed f16 (tiled, coalesced both sides)
    k_wtrans<<<dim3(32, 32),  256, 0, stream>>>(Wq, WqT, 1024, 1024);
    k_wtrans<<<dim3(32, 32),  256, 0, stream>>>(Wk, WkT, 1024, 1024);
    k_wtrans<<<dim3(32, 32),  256, 0, stream>>>(Wv, WvT, 1024, 1024);
    k_wtrans<<<dim3(32, 32),  256, 0, stream>>>(Wo, WoT, 1024, 1024);
    k_wtrans<<<dim3(128, 32), 256, 0, stream>>>(W1, W1T, 1024, 4096);
    k_wtrans<<<dim3(32, 128), 256, 0, stream>>>(W2, W2T, 4096, 1024);

    // 2) LN1
    k_layernorm<<<8192, 256, 0, stream>>>(x, ln1_g, ln1_b, Xn1);

    // 3) QKV projections (Q folds 1/sqrt(Dk)=0.125; V stored transposed)
    dim3 g1(1024 / 128, 8192 / 64);
    k_gemm<2><<<g1, 128, 0, stream>>>(Xn1, WqT, bq, nullptr, Qh, 8192, 1024, 1024, 0.125f);
    k_gemm<2><<<g1, 128, 0, stream>>>(Xn1, WkT, bk, nullptr, Kh, 8192, 1024, 1024, 1.0f);
    k_gemm<3><<<g1, 128, 0, stream>>>(Xn1, WvT, bv, nullptr, VtT, 8192, 1024, 1024, 1.0f);

    // 4) flash attention
    k_attn<<<dim3(2048 / 64, 64), 128, 0, stream>>>(Qh, Kh, VtT, Ctx);

    // 5) output projection + residual: X1 = x + Ctx @ Wo + bo
    k_gemm<0><<<g1, 128, 0, stream>>>(Ctx, WoT, bo, x, X1, 8192, 1024, 1024, 1.0f);

    // 6) LN2
    k_layernorm<<<8192, 256, 0, stream>>>(X1, ln2_g, ln2_b, Xn2);

    // 7) FFN: H = relu(Xn2 @ W1 + b1); out = X1 + H @ W2 + b2
    dim3 g2(4096 / 128, 8192 / 64);
    k_gemm<1><<<g2, 128, 0, stream>>>(Xn2, W1T, b1, nullptr, Hh, 8192, 4096, 1024, 1.0f);
    k_gemm<0><<<g1, 128, 0, stream>>>(Hh, W2T, b2, X1, d_out, 8192, 1024, 4096, 1.0f);
}
